// GraphAttn_51092930953222
// MI455X (gfx1250) — compile-verified, matrix operand order checked by reading, not run on hardware
//
#include <hip/hip_runtime.h>
#include <hip/hip_bf16.h>

// ---------------- problem constants (reference: B=8, N=1024, C=64, F=32, O=128)
#define C_IN   64
#define F_DIM  32
#define O_DIM  128
#define KW     3
#define KTOT   (KW * C_IN)          // 192
#define D_DIM  (O_DIM * F_DIM)      // 4096
#define ALPHA  0.2f

// ---------------- CDNA5 async global->LDS probe (fallback = manual copy)
#define HAVE_ASYNC_B128 0
#define HAVE_WAIT_ASYNC 0
#if defined(__has_builtin)
#  if __has_builtin(__builtin_amdgcn_global_load_async_to_lds_b128)
#    undef  HAVE_ASYNC_B128
#    define HAVE_ASYNC_B128 1
#  endif
#  if __has_builtin(__builtin_amdgcn_s_wait_asynccnt)
#    undef  HAVE_WAIT_ASYNC
#    define HAVE_WAIT_ASYNC 1
#  endif
#endif

// 4 x i32 vector pointers in explicit address spaces for the async builtin
typedef __attribute__((ext_vector_type(4))) int v4i;
typedef __attribute__((address_space(1))) v4i as1_v4i;
typedef __attribute__((address_space(3))) v4i as3_v4i;

// ---------------- WMMA types / fragment union
typedef __attribute__((ext_vector_type(16))) __bf16 v16bf;
typedef __attribute__((ext_vector_type(8)))  float  v8f;

union ABfrag {
    unsigned int u[8];
    v16bf        bf;
};

// float -> bf16 bits, round-to-nearest-even
static __device__ __forceinline__ unsigned short f2bf(float f) {
    unsigned int u = __float_as_uint(f);
    u += 0x7FFFu + ((u >> 16) & 1u);
    return (unsigned short)(u >> 16);
}

// ISA 7.12.2: 16-bit A/B fragment K-offset for VGPR v, lane-half hi
static __device__ __forceinline__ int frag_koff(int v, int hi) {
    return ((v & 4) << 2) + (hi << 3) + ((v & 3) << 1);
}

// =====================================================================
// Kernel 0: zero a1/a2 accumulators
// =====================================================================
__global__ void zero_kernel(float* __restrict__ p, int n) {
    int i = blockIdx.x * 256 + threadIdx.x;
    if (i < n) p[i] = 0.0f;
}

// =====================================================================
// Kernel 1: circular conv as WMMA GEMM  (one sample = one block, 8 waves)
//   h[o,f] = bias[o] + sum_{s,c} W[o][s*64+c] * x[c][(f+s-1) mod 32]
// also accumulates a[sample] = sum_d h[d]*attw[d]; optionally stores h (bf16).
// =====================================================================
__global__ __launch_bounds__(256)
void conv_wmma_kernel(const float* __restrict__ xin,    // [M, C, F]
                      const float* __restrict__ wconv,  // [O, C, 3]
                      const float* __restrict__ bias,   // [O]
                      const float* __restrict__ attw,   // [D]
                      unsigned short* __restrict__ hout,// [M, D] bf16
                      float* __restrict__ aout,         // [M]
                      int store_h)
{
    __shared__ __align__(16) unsigned short lds_w[O_DIM * KTOT];   // [o][kk], kk = s*64+c (48KB)
    __shared__ __align__(16) unsigned short lds_xT[F_DIM * C_IN];  // [f][c] transposed (4KB)

    const int t      = threadIdx.x;
    const int sample = blockIdx.x;

    // stage weights -> bf16 LDS, kk = s*64 + c ordering
    for (int idx = t; idx < O_DIM * KTOT; idx += 256) {
        int o  = idx / KTOT;
        int kk = idx - o * KTOT;
        int s  = kk >> 6;
        int c  = kk & 63;
        lds_w[idx] = f2bf(wconv[(o * C_IN + c) * KW + s]);
    }
    // stage x transposed -> lds_xT[f][c]  (contiguous c => 32-bit pair loads)
    const float* xs = xin + (size_t)sample * (C_IN * F_DIM);
    for (int idx = t; idx < C_IN * F_DIM; idx += 256) {
        int c = idx >> 5;
        int f = idx & 31;
        lds_xT[f * C_IN + c] = f2bf(xs[idx]);
    }
    __syncthreads();

    const int wave = t >> 5;       // M-strip: o rows [16*wave, 16*wave+16)
    const int lane = t & 31;
    const int m    = lane & 15;
    const int hi   = lane >> 4;

    v8f acc0 = {};                 // f tile 0..15
    v8f acc1 = {};                 // f tile 16..31

    for (int kc = 0; kc < 6; ++kc) {           // 6 x 32 = 192 deep
        const int s     = kc >> 1;
        const int cbase = (kc & 1) << 5;

        ABfrag afr, bfr0, bfr1;
        const int arow = (16 * wave + m) * KTOT + 32 * kc;
        const int f0   = (m + s + F_DIM - 1) & 31;        // circular shift
        const int f1   = (16 + m + s + F_DIM - 1) & 31;
#pragma unroll
        for (int v = 0; v < 8; ++v) {
            const int ko = frag_koff(v, hi);
            afr.u[v]  = *(const unsigned int*)&lds_w[arow + ko];
            bfr0.u[v] = *(const unsigned int*)&lds_xT[f0 * C_IN + cbase + ko];
            bfr1.u[v] = *(const unsigned int*)&lds_xT[f1 * C_IN + cbase + ko];
        }
        acc0 = __builtin_amdgcn_wmma_f32_16x16x32_bf16(false, afr.bf, false, bfr0.bf,
                                                       (short)0, acc0, false, false);
        acc1 = __builtin_amdgcn_wmma_f32_16x16x32_bf16(false, afr.bf, false, bfr1.bf,
                                                       (short)0, acc1, false, false);
    }

    // epilogue: bias, bf16 store, attention-weight dot product
    float apart = 0.0f;
    unsigned short* hb = hout + (size_t)sample * D_DIM;
#pragma unroll
    for (int r = 0; r < 8; ++r) {
        const int o  = 16 * wave + 8 * hi + r;   // C-matrix row mapping
        const float bz = bias[o];
        {
            float val = acc0[r] + bz;
            int d = o * F_DIM + m;
            apart += val * attw[d];
            if (store_h) hb[d] = f2bf(val);
        }
        {
            float val = acc1[r] + bz;
            int d = o * F_DIM + 16 + m;
            apart += val * attw[d];
            if (store_h) hb[d] = f2bf(val);
        }
    }
    // wave32 reduction then one atomic per wave
#pragma unroll
    for (int off = 16; off > 0; off >>= 1)
        apart += __shfl_down(apart, off, 32);
    if (lane == 0) atomicAdd(&aout[sample], apart);
}

// =====================================================================
// Kernel 2: per-column (j) online softmax over i, write bf16 att[b][i][j]
//   a[b,i,j] = lrelu(a1[b,i] + a2[b,j]); softmax over i (axis=1)
// =====================================================================
__global__ __launch_bounds__(256)
void att_kernel(const float* __restrict__ a1,     // [B,N]
                const float* __restrict__ a2,     // [B,N]
                unsigned short* __restrict__ att, // [B,N,N] bf16, row-major in i
                int N)
{
    const int b = blockIdx.y;
    const int j = blockIdx.x * 256 + threadIdx.x;
    const float* a1b = a1 + (size_t)b * N;
    const float  aj  = a2[(size_t)b * N + j];

    float mx = -3.4e38f, sum = 0.0f;
    for (int i = 0; i < N; ++i) {
        float v = a1b[i] + aj;
        v = (v >= 0.0f) ? v : ALPHA * v;
        float nm = fmaxf(mx, v);
        sum = sum * __expf(mx - nm) + __expf(v - nm);
        mx = nm;
    }
    const float inv = 1.0f / sum;

    unsigned short* attb = att + (size_t)b * N * N + j;
    for (int i = 0; i < N; ++i) {
        float v = a1b[i] + aj;
        v = (v >= 0.0f) ? v : ALPHA * v;
        attb[(size_t)i * N] = f2bf(__expf(v - mx) * inv);
    }
}

// =====================================================================
// Kernel 3: out[b] = att[b] (1024x1024) x h1[b] (1024x4096), bf16 WMMA, f32 out
//   block tile 128(i) x 64(d), K step 32, 8 waves in 4x2, 2x2 tiles per wave
//   A tile staged via GLOBAL_LOAD_ASYNC_TO_LDS_B128 when available
// =====================================================================
#define MBLK 128
#define NBLK 64
#define KSTEP 32

__global__ __launch_bounds__(256)
void attn_gemm_kernel(const unsigned short* __restrict__ att, // [B,N,N] bf16
                      const unsigned short* __restrict__ h1,  // [B,N,D] bf16
                      float* __restrict__ out,                // [B,N,D] f32
                      int N, int D)
{
    __shared__ __align__(16) unsigned short lds_a[MBLK * KSTEP]; // [i_local][k] (8KB)
    __shared__ __align__(16) unsigned short lds_b[NBLK * KSTEP]; // [d_local][k] (4KB)

    const int b  = blockIdx.z;
    const int i0 = blockIdx.y * MBLK;
    const int d0 = blockIdx.x * NBLK;
    const int t    = threadIdx.x;
    const int wave = t >> 5, lane = t & 31;
    const int wm = wave >> 1, wn = wave & 1;
    const int m  = lane & 15, hi = lane >> 4;

    const unsigned short* attB = att + (size_t)b * N * N;
    const unsigned short* h1B  = h1  + (size_t)b * N * D;

    v8f acc[2][2] = {};

    for (int k0 = 0; k0 < N; k0 += KSTEP) {
        __syncthreads();
        // stage A: att[i0..i0+127][k0..k0+31], 16B per chunk, 2 chunks/thread
        for (int u = t; u < (MBLK * KSTEP) / 8; u += 256) {
            int row = u >> 2;
            int kc8 = (u & 3) << 3;
#if HAVE_ASYNC_B128
            __builtin_amdgcn_global_load_async_to_lds_b128(
                (as1_v4i*)&attB[(size_t)(i0 + row) * N + k0 + kc8],
                (as3_v4i*)&lds_a[row * KSTEP + kc8],
                0, 0);
#else
            *(uint4*)&lds_a[row * KSTEP + kc8] =
                *(const uint4*)&attB[(size_t)(i0 + row) * N + k0 + kc8];
#endif
            if (k0 + KSTEP < N)   // hint next K tile -> global_prefetch_b8
                __builtin_prefetch(&attB[(size_t)(i0 + row) * N + k0 + KSTEP + kc8], 0, 1);
        }
        // stage B transposed: h1[k0+kk][d0+dd] -> lds_b[dd][kk] (contig k pairs)
        for (int u = t; u < (NBLK * KSTEP) / 8; u += 256) {
            int kk  = u >> 3;
            int dd8 = (u & 7) << 3;
            uint4 vv = *(const uint4*)&h1B[(size_t)(k0 + kk) * D + d0 + dd8];
            const unsigned short* pv = (const unsigned short*)&vv;
#pragma unroll
            for (int e = 0; e < 8; ++e)
                lds_b[(dd8 + e) * KSTEP + kk] = pv[e];
            if (k0 + KSTEP < N)
                __builtin_prefetch(&h1B[(size_t)(k0 + KSTEP + kk) * D + d0 + dd8], 0, 1);
        }
#if HAVE_ASYNC_B128
#  if HAVE_WAIT_ASYNC
        __builtin_amdgcn_s_wait_asynccnt(0);
#  else
        asm volatile("s_wait_asynccnt 0" ::: "memory");
#  endif
#endif
        __syncthreads();

        ABfrag afr[2], bfr[2];
#pragma unroll
        for (int tm = 0; tm < 2; ++tm) {
            const int row = 32 * wm + 16 * tm + m;
#pragma unroll
            for (int v = 0; v < 8; ++v)
                afr[tm].u[v] = *(const unsigned int*)&lds_a[row * KSTEP + frag_koff(v, hi)];
        }
#pragma unroll
        for (int tn = 0; tn < 2; ++tn) {
            const int col = 32 * wn + 16 * tn + m;
#pragma unroll
            for (int v = 0; v < 8; ++v)
                bfr[tn].u[v] = *(const unsigned int*)&lds_b[col * KSTEP + frag_koff(v, hi)];
        }
#pragma unroll
        for (int tm = 0; tm < 2; ++tm)
#pragma unroll
            for (int tn = 0; tn < 2; ++tn)
                acc[tm][tn] = __builtin_amdgcn_wmma_f32_16x16x32_bf16(
                    false, afr[tm].bf, false, bfr[tn].bf,
                    (short)0, acc[tm][tn], false, false);
    }

    // write f32 output per C-matrix layout
#pragma unroll
    for (int tm = 0; tm < 2; ++tm)
#pragma unroll
        for (int tn = 0; tn < 2; ++tn)
#pragma unroll
            for (int r = 0; r < 8; ++r) {
                int i = i0 + 32 * wm + 16 * tm + 8 * hi + r;
                int d = d0 + 32 * wn + 16 * tn + m;
                out[((size_t)b * N + i) * D + d] = acc[tm][tn][r];
            }
}

// =====================================================================
// launch
// =====================================================================
extern "C" void kernel_launch(void* const* d_in, const int* in_sizes, int n_in,
                              void* d_out, int out_size, void* d_ws, size_t ws_size,
                              hipStream_t stream) {
    const float* x      = (const float*)d_in[0];
    const float* nin    = (const float*)d_in[1];
    const float* conv_w = (const float*)d_in[2];
    const float* conv_b = (const float*)d_in[3];
    const float* att1_w = (const float*)d_in[4];
    const float* att2_w = (const float*)d_in[5];
    float* out = (float*)d_out;

    const int B = 8;
    const int N = 1024;
    const int M = B * N;            // 8192 samples
    const int D = D_DIM;            // 4096
    (void)in_sizes; (void)n_in; (void)out_size; (void)ws_size;

    // workspace layout: h1(bf16) | a1(f32) | a2(f32) | att(bf16)
    char* ws = (char*)d_ws;
    unsigned short* h1ws = (unsigned short*)ws;
    size_t off = (size_t)M * D * sizeof(unsigned short);          // 64 MB
    float* a1 = (float*)(ws + off); off += (size_t)M * sizeof(float);
    float* a2 = (float*)(ws + off); off += (size_t)M * sizeof(float);
    off = (off + 255) & ~(size_t)255;
    unsigned short* attws = (unsigned short*)(ws + off);          // 16 MB

    zero_kernel<<<(2 * M + 255) / 256, 256, 0, stream>>>(a1, 2 * M);
    // h1 + a1 (store h), then a2 only (h2 never needed downstream)
    conv_wmma_kernel<<<M, 256, 0, stream>>>(x,   conv_w, conv_b, att1_w, h1ws, a1, 1);
    conv_wmma_kernel<<<M, 256, 0, stream>>>(nin, conv_w, conv_b, att2_w, h1ws, a2, 0);
    att_kernel<<<dim3(N / 256, B), 256, 0, stream>>>(a1, a2, attws, N);
    attn_gemm_kernel<<<dim3(D / NBLK, N / MBLK, B), 256, 0, stream>>>(attws, h1ws, out, N, D);
}
